// MultiHeadSelfAttention_46196668236371
// MI455X (gfx1250) — matrix operand order repaired and checked by reading, not compiled, not run on hardware
//
#include <hip/hip_runtime.h>

// ---------------------------------------------------------------------------
// MHA forward for MI455X (gfx1250, wave32, WMMA).
// Precision: bf16 operands, f32 accumulation (compute-bound problem; bf16
// WMMA 16x16x32 gives 8x K-depth per instruction vs f32 WMMA 16x16x4).
// Data path: weights pre-transposed + V stored [b,h,d,s] so every WMMA
// B-operand is a contiguous 32B/lane load; whole bf16 working set is
// L2-resident (192MB), so we stream global->VGPR->WMMA.
// ---------------------------------------------------------------------------

typedef __bf16 bf16_t;
typedef __attribute__((ext_vector_type(16))) __bf16 v16bf;
typedef __attribute__((ext_vector_type(8)))  __bf16 v8bf;
typedef __attribute__((ext_vector_type(8)))  float  v8f;

#define B_  2
#define S_  2048
#define E_  2048
#define H_  16
#define D_  128
#define M_  (B_ * S_)          // 4096 rows for the projection GEMMs

static __device__ __forceinline__ v8f wmma_bf16(v16bf a, v16bf b, v8f c) {
  // (neg_a, A, neg_b, B, c_mod, C, reuse_a, reuse_b)
  return __builtin_amdgcn_wmma_f32_16x16x32_bf16(false, a, false, b,
                                                 (short)0, c, false, false);
}

static __device__ __forceinline__ v16bf join8(v8bf lo, v8bf hi) {
  union { v16bf v; v8bf h[2]; } u;
  u.h[0] = lo; u.h[1] = hi;
  return u.v;
}

static __device__ __forceinline__ v8f zero8() {
  v8f z;
#pragma unroll
  for (int i = 0; i < 8; ++i) z[i] = 0.f;
  return z;
}

// ---------------------------------------------------------------------------
// Elementwise converts
// ---------------------------------------------------------------------------
__global__ void f32_to_bf16_kernel(const float* __restrict__ in,
                                   bf16_t* __restrict__ out, int n) {
  for (int i = blockIdx.x * blockDim.x + threadIdx.x; i < n;
       i += gridDim.x * blockDim.x)
    out[i] = (bf16_t)in[i];
}

// out[n][k] = (bf16) in[k][n]  (E_ x E_) ; writes coalesced in k
__global__ void transpose_to_bf16_kernel(const float* __restrict__ in,
                                         bf16_t* __restrict__ out) {
  const int total = E_ * E_;
  for (int i = blockIdx.x * blockDim.x + threadIdx.x; i < total;
       i += gridDim.x * blockDim.x) {
    int nn = i / E_;
    int k  = i - nn * E_;
    out[(size_t)nn * E_ + k] = (bf16_t)in[(size_t)k * E_ + nn];
  }
}

// ---------------------------------------------------------------------------
// GEMM: Y[m][n] = sum_k X[m][k] * Wt[n][k]     (M_=4096, N=K=2048)
// Block: 256 threads = 8 waves. Wave w owns rows m0..m0+15, 64 cols.
// OUT_MODE 0: RoPE epilogue, write bf16 [b,h,s,d]   (Q, K)
// OUT_MODE 1: plain,         write bf16 [b,h,d,s]   (V transposed)
// OUT_MODE 2: plain,         write f32 row-major    (final projection)
// ---------------------------------------------------------------------------
template <int OUT_MODE>
__global__ __launch_bounds__(256) void gemm_wmma_kernel(
    const bf16_t* __restrict__ X, const bf16_t* __restrict__ Wt,
    void* __restrict__ Yout) {
  const int lane  = threadIdx.x & 31;
  const int wave  = threadIdx.x >> 5;
  const int col16 = lane & 15;
  const int hl    = lane >> 4;          // half of the wave (A k-half / C row-half)
  const int m0    = blockIdx.x * 128 + wave * 16;
  const int n0    = blockIdx.y * 64;

  v8f acc[4];
#pragma unroll
  for (int t = 0; t < 4; ++t) acc[t] = zero8();

  // A layout (16-bit, 16x32): lane row = col16, k offsets kb..kb+15 contiguous, kb = hl*16
  const bf16_t* aRow = X + (size_t)(m0 + col16) * E_ + hl * 16;
  // B layout: lane col = col16, contraction kk + hl*16 + [0..15] contiguous
  const bf16_t* bBase = Wt + (size_t)(n0 + col16) * E_ + hl * 16;

  for (int kk = 0; kk < E_; kk += 32) {
    v16bf afrag = join8(*(const v8bf*)(aRow + kk),
                        *(const v8bf*)(aRow + kk + 8));
#pragma unroll
    for (int t = 0; t < 4; ++t) {
      v16bf bfrag = *(const v16bf*)(bBase + (size_t)t * 16 * E_ + kk);
      acc[t] = wmma_bf16(afrag, bfrag, acc[t]);
    }
  }

  // ------------------------------ epilogue -------------------------------
#pragma unroll
  for (int t = 0; t < 4; ++t) {
    const int n = n0 + t * 16 + col16;       // output column (fixed per lane)
    const int h = n >> 7;                    // head
    const int d = n & 127;                   // dim within head
    float freq = 0.f, invf_arg;
    if (OUT_MODE == 0) {
      // inv_freq[t] = 10000^(-2t/128), t = d>>1 (same for both lanes of a pair)
      invf_arg = -(float)(d & ~1) * (9.210340371976184f / 128.0f); // ln(1e4)
      freq = __expf(invf_arg);
    }
#pragma unroll
    for (int j = 0; j < 8; ++j) {
      const int m = m0 + hl * 8 + j;         // output row (C layout)
      const int b = m >> 11;                 // / S_
      const int s = m & (S_ - 1);
      float c = acc[t][j];
      if (OUT_MODE == 0) {
        float partner = __shfl_xor(c, 1, 32);   // x0<->x1 (even/odd lane pair)
        float ang = (float)s * freq;
        float sn, cs;
        __sincosf(ang, &sn, &cs);
        float outv = (d & 1) ? (c * cs + partner * sn)   // x1*cos + x0*sin
                             : (c * cs - partner * sn);  // x0*cos - x1*sin
        ((bf16_t*)Yout)[((size_t)(b * H_ + h) * S_ + s) * D_ + d] = (bf16_t)outv;
      } else if (OUT_MODE == 1) {
        ((bf16_t*)Yout)[((size_t)(b * H_ + h) * D_ + d) * S_ + s] = (bf16_t)c;
      } else {
        ((float*)Yout)[(size_t)m * E_ + n] = c;
      }
    }
  }
}

// ---------------------------------------------------------------------------
// Flash attention, causal. Grid: (S_/128, B_*H_), block 256 = 8 waves.
// Wave w: 16 query rows q0..q0+15. Per 32-key tile: 8 WMMA (QK^T) +
// online softmax + LDS transpose of P + 8 WMMA (PV). No block barriers:
// the P buffer is wave-private and DS ops are in-order per wave.
// ---------------------------------------------------------------------------
__global__ __launch_bounds__(256) void attn_kernel(
    const bf16_t* __restrict__ Q,   // [b,h,s,d]
    const bf16_t* __restrict__ Kg,  // [b,h,s,d]
    const bf16_t* __restrict__ Vt,  // [b,h,d,s]
    bf16_t* __restrict__ ctx)       // [b,s,e]
{
  __shared__ __align__(32) bf16_t Pbuf[8][16][32];   // per-wave P tile

  const int lane  = threadIdx.x & 31;
  const int wave  = threadIdx.x >> 5;
  const int col16 = lane & 15;
  const int hl    = lane >> 4;
  const int bh    = blockIdx.y;
  const int q0    = blockIdx.x * 128 + wave * 16;

  const bf16_t* Qh = Q  + (size_t)bh * S_ * D_;
  const bf16_t* Kh = Kg + (size_t)bh * S_ * D_;
  const bf16_t* Vh = Vt + (size_t)bh * D_ * S_;

  // Q fragments: 4 contraction chunks of 32 over d
  v16bf qf[4];
  const bf16_t* qp = Qh + (size_t)(q0 + col16) * D_ + hl * 16;
#pragma unroll
  for (int c = 0; c < 4; ++c)
    qf[c] = join8(*(const v8bf*)(qp + 32 * c),
                  *(const v8bf*)(qp + 32 * c + 8));

  v8f o[8];
#pragma unroll
  for (int t = 0; t < 8; ++t) o[t] = zero8();
  float mrun[8], lrun[8];
#pragma unroll
  for (int j = 0; j < 8; ++j) { mrun[j] = -3.0e38f; lrun[j] = 0.f; }

  const float sm_scale = 0.088388347648318447f;      // 1/sqrt(128)
  const int nkt = ((q0 + 15) >> 5) + 1;              // causal 32-wide key tiles

  for (int kt = 0; kt < nkt; ++kt) {
    const int kp0 = kt * 32;

    // ---- scores: S = Q * K^T, two 16-col tiles ----
    v8f s0 = zero8(), s1 = zero8();
#pragma unroll
    for (int c = 0; c < 4; ++c) {
      v16bf bk0 = *(const v16bf*)(Kh + (size_t)(kp0 + col16) * D_ + 32 * c + hl * 16);
      v16bf bk1 = *(const v16bf*)(Kh + (size_t)(kp0 + 16 + col16) * D_ + 32 * c + hl * 16);
      s0 = wmma_bf16(qf[c], bk0, s0);
      s1 = wmma_bf16(qf[c], bk1, s1);
    }

    // ---- online softmax over the 32 columns ----
#pragma unroll
    for (int j = 0; j < 8; ++j) {
      const int qrow = q0 + hl * 8 + j;
      float a0 = s0[j] * sm_scale;
      float a1 = s1[j] * sm_scale;
      if (kp0 + col16 > qrow)       a0 = -3.0e38f;   // causal mask
      if (kp0 + 16 + col16 > qrow)  a1 = -3.0e38f;
      float rm = fmaxf(a0, a1);
      rm = fmaxf(rm, __shfl_xor(rm, 1, 32));
      rm = fmaxf(rm, __shfl_xor(rm, 2, 32));
      rm = fmaxf(rm, __shfl_xor(rm, 4, 32));
      rm = fmaxf(rm, __shfl_xor(rm, 8, 32));
      float mnew = fmaxf(mrun[j], rm);
      float sc   = __expf(mrun[j] - mnew);
      float e0   = __expf(a0 - mnew);
      float e1   = __expf(a1 - mnew);
      float rs = e0 + e1;
      rs += __shfl_xor(rs, 1, 32);
      rs += __shfl_xor(rs, 2, 32);
      rs += __shfl_xor(rs, 4, 32);
      rs += __shfl_xor(rs, 8, 32);
      lrun[j] = lrun[j] * sc + rs;
      mrun[j] = mnew;
#pragma unroll
      for (int t = 0; t < 8; ++t) o[t][j] *= sc;     // rescale accumulators
      Pbuf[wave][hl * 8 + j][col16]      = (bf16_t)e0;
      Pbuf[wave][hl * 8 + j][16 + col16] = (bf16_t)e1;
    }

    // DS is in-order per wave; force store->load ordering explicitly.
    asm volatile("s_wait_dscnt 0" ::: "memory");

    // ---- P tile in A layout (transpose via LDS) ----
    v16bf pa = join8(*(const v8bf*)&Pbuf[wave][col16][hl * 16],
                     *(const v8bf*)&Pbuf[wave][col16][hl * 16 + 8]);

    // ---- O += P * V ----
#pragma unroll
    for (int t = 0; t < 8; ++t) {
      v16bf bv = *(const v16bf*)(Vh + (size_t)(t * 16 + col16) * S_ + kp0 + hl * 16);
      o[t] = wmma_bf16(pa, bv, o[t]);
    }
  }

  // ---- normalize and write ctx[b, q, h*128 + d] ----
  const int b = bh >> 4;
  const int h = bh & 15;
#pragma unroll
  for (int j = 0; j < 8; ++j) {
    const int qrow = q0 + hl * 8 + j;
    const float inv = 1.0f / lrun[j];
    size_t base = ((size_t)(b * S_ + qrow)) * E_ + h * D_;
#pragma unroll
    for (int t = 0; t < 8; ++t)
      ctx[base + t * 16 + col16] = (bf16_t)(o[t][j] * inv);
  }
}

// ---------------------------------------------------------------------------
// Host launcher
// ---------------------------------------------------------------------------
extern "C" void kernel_launch(void* const* d_in, const int* in_sizes, int n_in,
                              void* d_out, int out_size, void* d_ws,
                              size_t ws_size, hipStream_t stream) {
  const float* x  = (const float*)d_in[0];
  // d_in[1] = start_pos (0, unused)
  const float* Wq = (const float*)d_in[2];
  const float* Wk = (const float*)d_in[3];
  const float* Wv = (const float*)d_in[4];
  const float* Wo = (const float*)d_in[5];
  float* out = (float*)d_out;

  // workspace carve-out (all bf16), 256B-aligned
  size_t off = 0;
  auto carve = [&](size_t elems) {
    bf16_t* p = (bf16_t*)((char*)d_ws + off);
    off = (off + elems * sizeof(bf16_t) + 255) & ~(size_t)255;
    return p;
  };
  bf16_t* xb   = carve((size_t)M_ * E_);
  bf16_t* wqt  = carve((size_t)E_ * E_);
  bf16_t* wkt  = carve((size_t)E_ * E_);
  bf16_t* wvt  = carve((size_t)E_ * E_);
  bf16_t* wot  = carve((size_t)E_ * E_);
  bf16_t* qb   = carve((size_t)M_ * E_);
  bf16_t* kb   = carve((size_t)M_ * E_);
  bf16_t* vtb  = carve((size_t)M_ * E_);
  bf16_t* ctxb = carve((size_t)M_ * E_);
  (void)ws_size; (void)in_sizes; (void)n_in; (void)out_size;

  // 1) converts / weight transposes
  f32_to_bf16_kernel<<<4096, 256, 0, stream>>>(x, xb, M_ * E_);
  transpose_to_bf16_kernel<<<4096, 256, 0, stream>>>(Wq, wqt);
  transpose_to_bf16_kernel<<<4096, 256, 0, stream>>>(Wk, wkt);
  transpose_to_bf16_kernel<<<4096, 256, 0, stream>>>(Wv, wvt);
  transpose_to_bf16_kernel<<<4096, 256, 0, stream>>>(Wo, wot);

  // 2) QKV projections (+ fused RoPE on Q,K; V stored transposed per head)
  dim3 gGemm(M_ / 128, E_ / 64);
  gemm_wmma_kernel<0><<<gGemm, 256, 0, stream>>>(xb, wqt, qb);
  gemm_wmma_kernel<0><<<gGemm, 256, 0, stream>>>(xb, wkt, kb);
  gemm_wmma_kernel<1><<<gGemm, 256, 0, stream>>>(xb, wvt, vtb);

  // 3) causal flash attention
  dim3 gAttn(S_ / 128, B_ * H_);
  attn_kernel<<<gAttn, 256, 0, stream>>>(qb, kb, vtb, ctxb);

  // 4) output projection -> fp32 d_out
  gemm_wmma_kernel<2><<<gGemm, 256, 0, stream>>>(ctxb, wot, out);
}
